// ContextualHistoryAdapterPrototype_53480932769816
// MI455X (gfx1250) — compile-verified, matrix operand order checked by reading, not run on hardware
//
#include <hip/hip_runtime.h>
#include <hip/hip_bf16.h>
#include <math.h>

typedef __attribute__((ext_vector_type(16))) _Float16 v16h;
typedef __attribute__((ext_vector_type(8)))  float    v8f;
typedef _Float16 h16;

#define B_   8
#define T_   128
#define D_   512
#define C_   1000
#define CP_  1024
#define L_   8
#define U_   16
#define E_   256
#define H_   4
#define DH_  128
#define QTOT 2048   // T_*U_

__device__ __forceinline__ float sigf(float x) { return 1.0f / (1.0f + __expf(-x)); }

// ---------------------------------------------------------------------------
// 16x32 f16 fragment load (A/B operand, NT layout), straight from f16 memory.
// lanes 0-15 -> row, K {kb..kb+7, kb+16..kb+23} with kb = (lane>=16)*8.
// Two 16-byte loads land directly in the fragment registers.
// ---------------------------------------------------------------------------
__device__ __forceinline__ v16h load_frag16(const h16* __restrict__ p) {
  v16h r;
  *(float4*)(&r)       = *(const float4*)(p);
  *(((float4*)(&r))+1) = *(const float4*)(p + 16);
  return r;
}

// C[M,N] = alpha * A[M,K](lda) @ W[N,K]^T(ldw).  M mult 16, N mult 64, K mult 32.
// One wave per 16x64 tile: 1 A frag x 4 W frags -> 4 WMMAs per K-step.
template <typename TO>
__global__ void __launch_bounds__(256) gemm_nt_wmma(
    const h16* __restrict__ A, const h16* __restrict__ W, TO* __restrict__ C,
    int M, int N, int K, int lda, int ldw, int ldc, float alpha)
{
  const int lane    = threadIdx.x & 31;
  const int wave    = threadIdx.x >> 5;
  const int tiles_n = N >> 6;
  const int tiles   = (M >> 4) * tiles_n;
  int tile = blockIdx.x * 8 + wave;
  if (tile >= tiles) return;
  const int tm   = tile / tiles_n;
  const int tn   = tile - tm * tiles_n;
  const int half = lane & 15;
  const int kb   = (lane >> 4) << 3;
  const h16* pa = A + (size_t)(tm * 16 + half) * lda + kb;
  const h16* pw = W + (size_t)(tn * 64 + half) * ldw + kb;
  const size_t wstep = (size_t)16 * ldw;
  v8f acc0 = {}, acc1 = {}, acc2 = {}, acc3 = {};
  for (int k0 = 0; k0 < K; k0 += 32) {
    v16h af = load_frag16(pa + k0);
    v16h w0 = load_frag16(pw + k0);
    v16h w1 = load_frag16(pw + wstep + k0);
    v16h w2 = load_frag16(pw + 2 * wstep + k0);
    v16h w3 = load_frag16(pw + 3 * wstep + k0);
    acc0 = __builtin_amdgcn_wmma_f32_16x16x32_f16(false, af, false, w0, (short)0, acc0, false, false);
    acc1 = __builtin_amdgcn_wmma_f32_16x16x32_f16(false, af, false, w1, (short)0, acc1, false, false);
    acc2 = __builtin_amdgcn_wmma_f32_16x16x32_f16(false, af, false, w2, (short)0, acc2, false, false);
    acc3 = __builtin_amdgcn_wmma_f32_16x16x32_f16(false, af, false, w3, (short)0, acc3, false, false);
  }
  // C/D layout: VGPR v -> row tm*16 + (lane>=16?8:0) + v, col 16j + (lane&15)
  const size_t r0 = (size_t)(tm * 16 + ((lane >> 4) << 3));
  v8f accs[4] = {acc0, acc1, acc2, acc3};
#pragma unroll
  for (int j = 0; j < 4; ++j) {
    TO* pc = C + r0 * ldc + tn * 64 + j * 16 + half;
#pragma unroll
    for (int v = 0; v < 8; ++v) pc[(size_t)v * ldc] = (TO)(alpha * accs[j][v]);
  }
}

// ---------------------------------------------------------------------------
__global__ void zero_f32(float* __restrict__ p, long n) {
  long i = (long)blockIdx.x * 256 + threadIdx.x;
  if (i < n) p[i] = 0.0f;
}
__global__ void zero_f16(h16* __restrict__ p, long n) {
  long i = (long)blockIdx.x * 256 + threadIdx.x;
  if (i < n) p[i] = (h16)0.0f;
}
__global__ void cvt_f32_f16(const float* __restrict__ in, h16* __restrict__ out, long n) {
  long i = (long)blockIdx.x * 256 + threadIdx.x;
  if (i < n) out[i] = (h16)in[i];
}

template <typename TI>
__global__ void gather_rows(const TI* __restrict__ table, const int* __restrict__ idx,
                            h16* __restrict__ out, int nrows, int width)
{
  long i = (long)blockIdx.x * 256 + threadIdx.x;
  if (i >= (long)nrows * width) return;
  int r = (int)(i / width);
  int e = (int)(i - (long)r * width);
  out[i] = (h16)table[(size_t)idx[r] * width + e];
}

// z = zih(step) + zr + bias ; gates i,f,g,o.  c stays f32, h written as f16.
__global__ void lstm_gate(const float* __restrict__ zih, long zih_stride,
                          const float* __restrict__ zr,
                          const float* __restrict__ bias,
                          float* __restrict__ cst, h16* __restrict__ hst,
                          h16* __restrict__ hout, long hout_stride,
                          int rows, int E)
{
  long i = (long)blockIdx.x * 256 + threadIdx.x;
  if (i >= (long)rows * E) return;
  int r = (int)(i / E);
  int e = (int)(i - (long)r * E);
  const float* z1 = zih + (size_t)r * zih_stride;
  const float* z2 = zr + (size_t)r * 4 * E;
  float gi = z1[e]         + z2[e]         + bias[e];
  float gf = z1[E + e]     + z2[E + e]     + bias[E + e];
  float gg = z1[2 * E + e] + z2[2 * E + e] + bias[2 * E + e];
  float go = z1[3 * E + e] + z2[3 * E + e] + bias[3 * E + e];
  float c  = sigf(gf) * cst[i] + sigf(gi) * tanhf(gg);
  float h  = sigf(go) * tanhf(c);
  cst[i] = c;
  h16 hh = (h16)h;
  hst[i] = hh;
  hout[(size_t)r * hout_stride + e] = hh;
}

__global__ void concat2_f16(const h16* __restrict__ a, const h16* __restrict__ b,
                            h16* __restrict__ out, int rows, int w)
{
  long i = (long)blockIdx.x * 256 + threadIdx.x;
  if (i >= (long)rows * 2 * w) return;
  int r = (int)(i / (2 * w));
  int c = (int)(i - (long)r * 2 * w);
  out[i] = (c < w) ? a[(size_t)r * w + c] : b[(size_t)r * w + c - w];
}

__global__ void ctx_mean_k(const float* __restrict__ seq, const int* __restrict__ ilens,
                           h16* __restrict__ out)
{
  long i = (long)blockIdx.x * 256 + threadIdx.x;
  if (i >= (long)CP_ * D_) return;
  int c = (int)(i / D_);
  int d = (int)(i - (long)c * D_);
  if (c >= C_) { out[i] = (h16)0.0f; return; }
  int n = ilens[c];
  float s = 0.0f;
  for (int l = 0; l < n && l < L_; ++l) s += seq[((size_t)c * L_ + l) * D_ + d];
  out[i] = (h16)(s / (float)n);
}

__global__ void fuse_k(const float* __restrict__ mp, const float* __restrict__ hist,
                       h16* __restrict__ fused)
{
  long i = (long)blockIdx.x * 256 + threadIdx.x;
  if (i >= (long)B_ * QTOT * D_) return;
  int d = (int)(i & (D_ - 1));
  long r = i >> 9;
  int u = (int)(r & (U_ - 1));
  int t = (int)((r >> 4) & (T_ - 1));
  int b = (int)(r >> 11);
  fused[i] = (h16)(mp[((size_t)(b * T_ + t)) * D_ + d] +
                   hist[((size_t)(b * U_ + u)) * D_ + d]);
}

__global__ void transpose_f16(const h16* __restrict__ src, h16* __restrict__ dst,
                              int rows, int cols)
{
  long i = (long)blockIdx.x * 256 + threadIdx.x;
  if (i >= (long)rows * cols) return;
  int r = (int)(i / cols);
  int c = (int)(i - (long)r * cols);
  dst[(size_t)c * rows + r] = src[i];
}

// One block per query row: softmax over C_=1000 of scores (ld CP_), writes
// f16 attn (zero-padded to CP_), accumulates logit/prob (mean over H).
__global__ void __launch_bounds__(256) softmax_k(
    const float* __restrict__ scores, h16* __restrict__ attn,
    float* __restrict__ logit, float* __restrict__ prob)
{
  __shared__ float red[256];
  int q = blockIdx.x;
  const float* srow = scores + (size_t)q * CP_;
  float m = -1e30f;
  for (int c = threadIdx.x; c < C_; c += 256) m = fmaxf(m, srow[c]);
  red[threadIdx.x] = m; __syncthreads();
  for (int s = 128; s > 0; s >>= 1) {
    if (threadIdx.x < s) red[threadIdx.x] = fmaxf(red[threadIdx.x], red[threadIdx.x + s]);
    __syncthreads();
  }
  m = red[0]; __syncthreads();
  float sum = 0.0f;
  for (int c = threadIdx.x; c < C_; c += 256) sum += __expf(srow[c] - m);
  red[threadIdx.x] = sum; __syncthreads();
  for (int s = 128; s > 0; s >>= 1) {
    if (threadIdx.x < s) red[threadIdx.x] += red[threadIdx.x + s];
    __syncthreads();
  }
  float inv = 1.0f / red[0];
  h16*   arow = attn  + (size_t)q * CP_;
  float* lrow = logit + (size_t)q * C_;
  float* prow = prob  + (size_t)q * C_;
  for (int c = threadIdx.x; c < CP_; c += 256) {
    if (c < C_) {
      float s0 = srow[c];
      float a  = __expf(s0 - m) * inv;
      arow[c] = (h16)a;
      lrow[c] += 0.25f * s0;
      prow[c] += 0.25f * a;
    } else {
      arow[c] = (h16)0.0f;
    }
  }
}

// Sequential monotonic scan over T. One wave per batch element.
__global__ void scan_k(const float* __restrict__ prob, const float* __restrict__ out2,
                       const int* __restrict__ hidx, float* __restrict__ o)
{
  int b    = threadIdx.x >> 5;
  int lane = threadIdx.x & 31;
  if (b >= B_) return;
  int u = 0;
  for (int t = 0; t < T_; ++t) {
    int uc  = (u < U_ - 1) ? u : (U_ - 1);
    int vu  = (u + 1 < U_ - 1) ? (u + 1) : (U_ - 1);
    int tok = hidx[b * U_ + vu];
    const float* p = prob + (((size_t)b * T_ + t) * U_ + uc) * C_;
    float blank = p[0];
    float tp    = p[tok];
    const float* src = out2 + ((size_t)b * QTOT + (size_t)t * U_ + uc) * D_;
    float* dst = o + ((size_t)b * T_ + t) * D_;
    for (int d = lane; d < D_; d += 32) dst[d] = src[d];
    u = uc + ((tp > blank) ? 1 : 0);
  }
}

// ---------------------------------------------------------------------------
extern "C" void kernel_launch(void* const* d_in, const int* in_sizes, int n_in,
                              void* d_out, int out_size, void* d_ws, size_t ws_size,
                              hipStream_t stream)
{
  (void)in_sizes; (void)n_in; (void)out_size; (void)ws_size;
  const float* model_embed = (const float*)d_in[0];
  const float* emb_table   = (const float*)d_in[1];
  const float* ctx_b_f     = (const float*)d_in[4];
  const float* ctx_b_b     = (const float*)d_in[7];
  const float* hist_b      = (const float*)d_in[11];
  const int*   ctx_idxs    = (const int*)d_in[18];
  const int*   hist_idx    = (const int*)d_in[19];
  const int*   ctx_ilens   = (const int*)d_in[20];

  float* out_o     = (float*)d_out;                          // (B,T,D)
  float* out_logit = out_o + (size_t)B_ * T_ * D_;           // (B,T,U,C)
  float* out_prob  = out_logit + (size_t)B_ * T_ * U_ * C_;  // (B,T,U,C)

  char* cur = (char*)d_ws;
  auto allocF = [&](size_t n) { float* p = (float*)cur; cur += ((n * 4 + 255) & ~(size_t)255); return p; };
  auto allocH = [&](size_t n) { h16*   p = (h16*)cur;   cur += ((n * 2 + 255) & ~(size_t)255); return p; };

  // ---- f32 scratch ----
  float* zih_f  = allocF((size_t)C_ * L_ * 4 * E_);   // (8000,1024)
  float* zih_b  = allocF((size_t)C_ * L_ * 4 * E_);
  float* cstf   = allocF((size_t)2 * CP_ * E_);       // cF, cB
  float* cstb   = cstf + (size_t)CP_ * E_;
  float* zr     = allocF((size_t)CP_ * 4 * E_);       // (1024,1024)
  float* seq    = allocF((size_t)C_ * L_ * D_);       // (8000,512)
  float* zih_h  = allocF((size_t)B_ * U_ * 4 * D_);   // (128,2048)
  float* zr_h   = allocF((size_t)16 * 4 * D_);        // (16,2048)
  float* c_h    = allocF((size_t)16 * D_);
  float* hist   = allocF((size_t)B_ * U_ * D_);       // (128,512)
  float* mp     = allocF((size_t)B_ * T_ * D_);       // (1024,512)
  float* sc_s   = allocF((size_t)QTOT * CP_);         // per-(b,h) scores
  float* out2   = allocF((size_t)B_ * QTOT * D_);     // (16384,512)

  // ---- f16 weights (converted once per launch) ----
  h16* wih_f16 = allocH((size_t)4 * E_ * E_);         // (1024,256)
  h16* whh_f16 = allocH((size_t)4 * E_ * E_);
  h16* wih_b16 = allocH((size_t)4 * E_ * E_);
  h16* whh_b16 = allocH((size_t)4 * E_ * E_);
  h16* cow16   = allocH((size_t)D_ * 2 * E_);         // (512,512)
  h16* hwih16  = allocH((size_t)4 * D_ * D_);         // (2048,512)
  h16* hwhh16  = allocH((size_t)4 * D_ * D_);
  h16* how16   = allocH((size_t)D_ * D_);
  h16* wq16    = allocH((size_t)D_ * D_);
  h16* wk16    = allocH((size_t)D_ * D_);
  h16* wv16    = allocH((size_t)D_ * D_);
  h16* wo16    = allocH((size_t)D_ * D_);
  h16* aw16    = allocH((size_t)D_ * D_);

  // ---- f16 activations ----
  h16* emb16    = allocH((size_t)C_ * L_ * E_);       // (8000,256)
  h16* hf_all16 = allocH((size_t)C_ * L_ * E_);
  h16* hb_all16 = allocH((size_t)C_ * L_ * E_);
  h16* hstf16   = allocH((size_t)2 * CP_ * E_);       // hF, hB states (padded)
  h16* hstb16   = hstf16 + (size_t)CP_ * E_;
  h16* cat16    = allocH((size_t)C_ * L_ * 2 * E_);   // (8000,512)
  h16* cmean16  = allocH((size_t)CP_ * D_);           // (1024,512) padded
  h16* histin16 = allocH((size_t)B_ * U_ * D_);       // (128,512)
  h16* h_h16    = allocH((size_t)16 * D_);
  h16* histall16= allocH((size_t)B_ * U_ * D_);
  h16* me16     = allocH((size_t)B_ * T_ * D_);       // model_embed f16
  h16* fused16  = allocH((size_t)B_ * QTOT * D_);     // (16384,512)
  h16* q16      = allocH((size_t)B_ * QTOT * D_);
  h16* k16      = allocH((size_t)CP_ * D_);
  h16* v16      = allocH((size_t)CP_ * D_);
  h16* vT16     = allocH((size_t)D_ * CP_);
  h16* attn16   = allocH((size_t)QTOT * CP_);
  h16* outb16   = allocH((size_t)B_ * QTOT * D_);

  auto g1 = [](long n) { return dim3((unsigned)((n + 255) / 256)); };
  auto gemmF = [&](const h16* A, const h16* W, float* C, int M, int N, int K,
                   int lda, int ldw, int ldc, float alpha) {
    int tiles = (M / 16) * (N / 64);
    gemm_nt_wmma<float><<<dim3((unsigned)((tiles + 7) / 8)), dim3(256), 0, stream>>>(
        A, W, C, M, N, K, lda, ldw, ldc, alpha);
  };
  auto gemmH = [&](const h16* A, const h16* W, h16* C, int M, int N, int K,
                   int lda, int ldw, int ldc, float alpha) {
    int tiles = (M / 16) * (N / 64);
    gemm_nt_wmma<h16><<<dim3((unsigned)((tiles + 7) / 8)), dim3(256), 0, stream>>>(
        A, W, C, M, N, K, lda, ldw, ldc, alpha);
  };
  auto cvt = [&](const float* in, h16* out, long n) {
    cvt_f32_f16<<<g1(n), 256, 0, stream>>>(in, out, n);
  };

  // --- deterministic init ---------------------------------------------------
  zero_f32<<<g1((long)2 * CP_ * E_), 256, 0, stream>>>(cstf, (long)2 * CP_ * E_);
  zero_f32<<<g1((long)16 * D_), 256, 0, stream>>>(c_h, (long)16 * D_);
  zero_f16<<<g1((long)2 * CP_ * E_), 256, 0, stream>>>(hstf16, (long)2 * CP_ * E_);
  zero_f16<<<g1((long)16 * D_), 256, 0, stream>>>(h_h16, (long)16 * D_);
  zero_f32<<<g1((long)2 * B_ * T_ * U_ * C_), 256, 0, stream>>>(
      out_logit, (long)2 * B_ * T_ * U_ * C_);

  // --- weight conversion ----------------------------------------------------
  cvt((const float*)d_in[2],  wih_f16, (long)4 * E_ * E_);
  cvt((const float*)d_in[3],  whh_f16, (long)4 * E_ * E_);
  cvt((const float*)d_in[5],  wih_b16, (long)4 * E_ * E_);
  cvt((const float*)d_in[6],  whh_b16, (long)4 * E_ * E_);
  cvt((const float*)d_in[8],  cow16,   (long)D_ * 2 * E_);
  cvt((const float*)d_in[9],  hwih16,  (long)4 * D_ * D_);
  cvt((const float*)d_in[10], hwhh16,  (long)4 * D_ * D_);
  cvt((const float*)d_in[12], how16,   (long)D_ * D_);
  cvt((const float*)d_in[13], wq16,    (long)D_ * D_);
  cvt((const float*)d_in[14], wk16,    (long)D_ * D_);
  cvt((const float*)d_in[15], wv16,    (long)D_ * D_);
  cvt((const float*)d_in[16], wo16,    (long)D_ * D_);
  cvt((const float*)d_in[17], aw16,    (long)D_ * D_);
  cvt(model_embed, me16, (long)B_ * T_ * D_);

  // --- 1. embedding gather --------------------------------------------------
  gather_rows<float><<<g1((long)C_ * L_ * E_), 256, 0, stream>>>(
      emb_table, ctx_idxs, emb16, C_ * L_, E_);

  // --- 2. x @ Wih^T for all timesteps ---------------------------------------
  gemmF(emb16, wih_f16, zih_f, C_ * L_, 4 * E_, E_, E_, E_, 4 * E_, 1.0f);
  gemmF(emb16, wih_b16, zih_b, C_ * L_, 4 * E_, E_, E_, E_, 4 * E_, 1.0f);

  // --- 3. context BiLSTM ----------------------------------------------------
  for (int t = 0; t < L_; ++t) {
    gemmF(hstf16, whh_f16, zr, CP_, 4 * E_, E_, E_, E_, 4 * E_, 1.0f);
    lstm_gate<<<g1((long)C_ * E_), 256, 0, stream>>>(
        zih_f + (size_t)t * 4 * E_, (long)L_ * 4 * E_, zr, ctx_b_f,
        cstf, hstf16, hf_all16 + (size_t)t * E_, (long)L_ * E_, C_, E_);
  }
  for (int s = 0; s < L_; ++s) {
    int t = L_ - 1 - s;
    gemmF(hstb16, whh_b16, zr, CP_, 4 * E_, E_, E_, E_, 4 * E_, 1.0f);
    lstm_gate<<<g1((long)C_ * E_), 256, 0, stream>>>(
        zih_b + (size_t)t * 4 * E_, (long)L_ * 4 * E_, zr, ctx_b_b,
        cstb, hstb16, hb_all16 + (size_t)t * E_, (long)L_ * E_, C_, E_);
  }

  // --- 4. concat + projection + masked mean ---------------------------------
  concat2_f16<<<g1((long)C_ * L_ * 2 * E_), 256, 0, stream>>>(
      hf_all16, hb_all16, cat16, C_ * L_, E_);
  gemmF(cat16, cow16, seq, C_ * L_, D_, 2 * E_, 2 * E_, 2 * E_, D_, 1.0f);
  ctx_mean_k<<<g1((long)CP_ * D_), 256, 0, stream>>>(seq, ctx_ilens, cmean16);

  // --- 5. history LSTM ------------------------------------------------------
  gather_rows<h16><<<g1((long)B_ * U_ * D_), 256, 0, stream>>>(
      cmean16, hist_idx, histin16, B_ * U_, D_);
  gemmF(histin16, hwih16, zih_h, B_ * U_, 4 * D_, D_, D_, D_, 4 * D_, 1.0f);
  for (int t = 0; t < U_; ++t) {
    gemmF(h_h16, hwhh16, zr_h, 16, 4 * D_, D_, D_, D_, 4 * D_, 1.0f);
    lstm_gate<<<g1((long)B_ * D_), 256, 0, stream>>>(
        zih_h + (size_t)t * 4 * D_, (long)U_ * 4 * D_, zr_h, hist_b,
        c_h, h_h16, histall16 + (size_t)t * D_, (long)U_ * D_, B_, D_);
  }
  gemmF(histall16, how16, hist, B_ * U_, D_, D_, D_, D_, D_, 1.0f);

  // --- 6. acoustic projection, fuse, q/k/v ----------------------------------
  gemmF(me16, aw16, mp, B_ * T_, D_, D_, D_, D_, D_, 1.0f);
  fuse_k<<<g1((long)B_ * QTOT * D_), 256, 0, stream>>>(mp, hist, fused16);
  gemmH(fused16, wq16, q16, B_ * QTOT, D_, D_, D_, D_, D_, 1.0f);
  gemmH(cmean16, wk16, k16, CP_, D_, D_, D_, D_, D_, 1.0f);
  gemmH(cmean16, wv16, v16, CP_, D_, D_, D_, D_, D_, 1.0f);
  transpose_f16<<<g1((long)CP_ * D_), 256, 0, stream>>>(v16, vT16, CP_, D_);

  // --- 7. attention per (b, h) ----------------------------------------------
  const float inv_sqrt_dh = 0.08838834764831845f;  // 1/sqrt(128)
  for (int b = 0; b < B_; ++b) {
    for (int h = 0; h < H_; ++h) {
      gemmF(q16 + (size_t)b * QTOT * D_ + h * DH_, k16 + h * DH_, sc_s,
            QTOT, CP_, DH_, D_, D_, CP_, inv_sqrt_dh);
      softmax_k<<<dim3(QTOT), dim3(256), 0, stream>>>(
          sc_s, attn16,
          out_logit + (size_t)b * QTOT * C_, out_prob + (size_t)b * QTOT * C_);
      gemmH(attn16, vT16 + (size_t)h * DH_ * CP_,
            outb16 + (size_t)b * QTOT * D_ + h * DH_,
            QTOT, DH_, CP_, CP_, CP_, D_, 1.0f);
    }
  }

  // --- 8. output projection + sequential scan -------------------------------
  gemmF(outb16, wo16, out2, B_ * QTOT, D_, D_, D_, D_, D_, 1.0f);
  scan_k<<<dim3(1), dim3(256), 0, stream>>>(out_prob, out2, hist_idx, out_o);
}
